// NystromformerSelfAttention_15977278341630
// MI455X (gfx1250) — matrix operand order repaired and checked by reading, not compiled, not run on hardware
//
#include <hip/hip_runtime.h>
#include <hip/hip_bf16.h>
#include <math.h>

#define BB 4
#define SS 4096
#define HH 16
#define DD 64
#define LL 64
#define DM 1024
#define CKSZ 65

typedef __attribute__((ext_vector_type(16))) __bf16 bf16x16;
typedef __attribute__((ext_vector_type(8)))  __bf16 bf16x8;
typedef __attribute__((ext_vector_type(8)))  float  f32x8;
typedef __attribute__((ext_vector_type(2)))  float  f32x2;

__device__ __forceinline__ f32x8 wmma_bf16(bf16x16 a, bf16x16 b, f32x8 c) {
  return __builtin_amdgcn_wmma_f32_16x16x32_bf16(false, a, false, b, (short)0, c, false, false);
}
__device__ __forceinline__ f32x8 wmma_f32k4(f32x2 a, f32x2 b, f32x8 c) {
  return __builtin_amdgcn_wmma_f32_16x16x4_f32(false, a, false, b, (short)0, c, false, false);
}

__device__ __forceinline__ f32x8 zero8() {
  f32x8 v;
#pragma unroll
  for (int i = 0; i < 8; ++i) v[i] = 0.f;
  return v;
}

__device__ __forceinline__ bf16x16 frag_pair(const __bf16* p0, const __bf16* p1) {
  union { bf16x16 v; bf16x8 h[2]; } u;
  u.h[0] = *reinterpret_cast<const bf16x8*>(p0);
  u.h[1] = *reinterpret_cast<const bf16x8*>(p1);
  return u.v;
}
// A fragment (16x32 bf16): row-major src, 32-wide K window at kb. 'row' already includes lane&15.
// elem i(0..7) <-> K = kb + hi*8 + i ; elem 8+i <-> K = kb + 16 + hi*8 + i
__device__ __forceinline__ bf16x16 frag_a(const __bf16* src, int row, int ld, int kb, int hi) {
  const __bf16* p = src + (size_t)row * ld + kb + hi * 8;
  return frag_pair(p, p + 16);
}
// B fragment (32x16 bf16): K-contiguous per column: src laid out [n][ldk].
// elem i(0..7) <-> K = kb + hi*8 + i ; elem 8+i <-> K = kb + 16 + hi*8 + i, N = n
__device__ __forceinline__ bf16x16 frag_b(const __bf16* src, int n, int ldk, int kb, int hi) {
  const __bf16* p = src + (size_t)n * ldk + kb + hi * 8;
  return frag_pair(p, p + 16);
}

__device__ __forceinline__ float redmax16(float v) {
  v = fmaxf(v, __shfl_xor(v, 8, 32));
  v = fmaxf(v, __shfl_xor(v, 4, 32));
  v = fmaxf(v, __shfl_xor(v, 2, 32));
  v = fmaxf(v, __shfl_xor(v, 1, 32));
  return v;
}
__device__ __forceinline__ float redsum16(float v) {
  v += __shfl_xor(v, 8, 32);
  v += __shfl_xor(v, 4, 32);
  v += __shfl_xor(v, 2, 32);
  v += __shfl_xor(v, 1, 32);
  return v;
}

// Issue one per-lane 16B async copy global->LDS (ASYNCcnt-tracked, no VGPR data).
__device__ __forceinline__ void async_cp16(const __bf16* g, __bf16* l) {
  unsigned loff = (unsigned)(uintptr_t)l;  // LDS aperture: low 32 bits = LDS offset
  asm volatile("global_load_async_to_lds_b128 %0, %1, off"
               :: "v"(loff), "v"(g) : "memory");
}

__global__ void k_zero_scalar(unsigned* p) { if (threadIdx.x == 0) *p = 0u; }

// ---------------- prepass: X f32 -> bf16 -------------------------------------
__global__ __launch_bounds__(256)
void k_cvt_x(const float* __restrict__ X, __bf16* __restrict__ Xb) {
  size_t idx = ((size_t)blockIdx.x * 256 + threadIdx.x) * 8;
  const float4 a = *reinterpret_cast<const float4*>(X + idx);
  const float4 b = *reinterpret_cast<const float4*>(X + idx + 4);
  union { bf16x8 v; __bf16 e[8]; } u;
  u.e[0] = (__bf16)a.x; u.e[1] = (__bf16)a.y; u.e[2] = (__bf16)a.z; u.e[3] = (__bf16)a.w;
  u.e[4] = (__bf16)b.x; u.e[5] = (__bf16)b.y; u.e[6] = (__bf16)b.z; u.e[7] = (__bf16)b.w;
  *reinterpret_cast<bf16x8*>(Xb + idx) = u.v;
}

// ---------------- prepass: W f32 [k][n] -> bf16 transposed [n][k] ------------
__global__ __launch_bounds__(256)
void k_cvt_wt(const float* __restrict__ Wq, const float* __restrict__ Wk,
              const float* __restrict__ Wv, __bf16* __restrict__ Wt) {
  const int z = blockIdx.z;
  const float* W = (z == 0) ? Wq : (z == 1) ? Wk : Wv;
  __bf16* dst = Wt + (size_t)z * DM * DM;
  const int k0 = blockIdx.x * 32, n0 = blockIdx.y * 32;
  __shared__ float t[32][33];
  const int tid = threadIdx.x;
  const int c = tid & 31, r8 = tid >> 5;
#pragma unroll
  for (int i = 0; i < 4; ++i) {
    int r = r8 + i * 8;
    t[r][c] = W[(size_t)(k0 + r) * DM + n0 + c];
  }
  __syncthreads();
#pragma unroll
  for (int i = 0; i < 4; ++i) {
    int r = r8 + i * 8;
    dst[(size_t)(n0 + r) * DM + k0 + c] = (__bf16)t[c][r];
  }
}

// ---------------- QKV GEMM: bf16 WMMA, async-to-LDS double-buffered ----------
__global__ __launch_bounds__(256)
void k_qkv(const __bf16* __restrict__ Xb, const __bf16* __restrict__ Wt,
           const float* __restrict__ bq, const float* __restrict__ bk,
           const float* __restrict__ bv,
           __bf16* __restrict__ qb, __bf16* __restrict__ kbuf,
           __bf16* __restrict__ vT) {
  const int which = blockIdx.z;
  const __bf16* Wsrc = Wt + (size_t)which * DM * DM;          // [n][k] bf16
  const float* bias = (which == 0) ? bq : (which == 1) ? bk : bv;
  const int n0 = blockIdx.x * 128;
  const int m0 = blockIdx.y * 128;
  __shared__ __align__(16) __bf16 lA[2][128 * 32];   // [m][k]
  __shared__ __align__(16) __bf16 lB[2][128 * 32];   // [n][k]
  const int tid = threadIdx.x;
  const int lane = tid & 31, wid = tid >> 5;
  const int lo = lane & 15, hi = lane >> 4;
  const int mW = (wid & 1) * 64, nW = (wid >> 1) * 32;

  f32x8 acc[4][2];
#pragma unroll
  for (int mt = 0; mt < 4; ++mt)
#pragma unroll
    for (int nt = 0; nt < 2; ++nt) acc[mt][nt] = zero8();

  // per stage: each wave issues 4 async b128 copies (2 A + 2 B)
  const int c0 = tid, c1 = tid + 256;       // 512 x 16B chunks per 8KB tile
  const int rA0 = c0 >> 2, kA0 = (c0 & 3) * 8;
  const int rA1 = c1 >> 2, kA1 = (c1 & 3) * 8;

#define ISSUE_STAGE(kb, buf)                                                     \
  do {                                                                           \
    async_cp16(Xb   + (size_t)(m0 + rA0) * DM + (kb) + kA0, &lA[buf][c0 * 8]);   \
    async_cp16(Xb   + (size_t)(m0 + rA1) * DM + (kb) + kA1, &lA[buf][c1 * 8]);   \
    async_cp16(Wsrc + (size_t)(n0 + rA0) * DM + (kb) + kA0, &lB[buf][c0 * 8]);   \
    async_cp16(Wsrc + (size_t)(n0 + rA1) * DM + (kb) + kA1, &lB[buf][c1 * 8]);   \
  } while (0)

  ISSUE_STAGE(0, 0);
  for (int kt = 0; kt < 32; ++kt) {
    const int buf = kt & 1;
    if (kt < 31) {
      ISSUE_STAGE((kt + 1) * 32, buf ^ 1);
      asm volatile("s_wait_asynccnt 0x4" ::: "memory");   // drain stage kt
    } else {
      asm volatile("s_wait_asynccnt 0x0" ::: "memory");
    }
    __syncthreads();
    bf16x16 bfr[2];
#pragma unroll
    for (int nt = 0; nt < 2; ++nt)
      bfr[nt] = frag_b(lB[buf], nW + nt * 16 + lo, 32, 0, hi);
#pragma unroll
    for (int mt = 0; mt < 4; ++mt) {
      bf16x16 afr = frag_a(lA[buf], mW + mt * 16 + lo, 32, 0, hi);
#pragma unroll
      for (int nt = 0; nt < 2; ++nt)
        acc[mt][nt] = wmma_bf16(afr, bfr[nt], acc[mt][nt]);
    }
    __syncthreads();   // all reads done before stage kt+2 overwrites this buffer
  }
#undef ISSUE_STAGE

  const float scale = 0.35355339059327373f;  // 64^-0.25
#pragma unroll
  for (int mt = 0; mt < 4; ++mt)
#pragma unroll
    for (int nt = 0; nt < 2; ++nt)
#pragma unroll
      for (int r = 0; r < 8; ++r) {
        int m = m0 + mW + mt * 16 + r + hi * 8;
        int n = n0 + nW + nt * 16 + lo;
        float v = acc[mt][nt][r] + bias[n];
        int bidx = m >> 12, s = m & 4095;
        int h = n >> 6, d = n & 63;
        int bh = bidx * HH + h;
        if (which == 0)      qb[((size_t)bh * SS + s) * DD + d]   = (__bf16)(v * scale);
        else if (which == 1) kbuf[((size_t)bh * SS + s) * DD + d] = (__bf16)(v * scale);
        else                 vT[((size_t)bh * DD + d) * SS + s]   = (__bf16)v;
      }
}

// ---------------- landmark means ---------------------------------------------
__global__ void k_landmarks(const __bf16* __restrict__ qb, const __bf16* __restrict__ kbuf,
                            __bf16* __restrict__ ql, __bf16* __restrict__ kl) {
  int bh = blockIdx.x >> 6, l = blockIdx.x & 63, d = threadIdx.x;
  const __bf16* src = qb; __bf16* dst = ql;
#pragma unroll
  for (int a = 0; a < 2; ++a) {
    if (a == 1) { src = kbuf; dst = kl; }
    float s = 0.f;
    const __bf16* p = src + ((size_t)bh * SS + l * 64) * DD + d;
    for (int r = 0; r < 64; ++r) s += (float)p[r * DD];
    dst[((size_t)bh * LL + l) * DD + d] = (__bf16)(s * (1.f / 64.f));
  }
}

// ---------------- kernel_2 = softmax(q_l k_l^T), + global max colsum ---------
__global__ __launch_bounds__(128)
void k_kernel2(const __bf16* __restrict__ ql, const __bf16* __restrict__ kl,
               float* __restrict__ k2, unsigned* __restrict__ gmax) {
  const int bh = blockIdx.x;
  const int tid = threadIdx.x, lane = tid & 31, wid = tid >> 5;
  const int lo = lane & 15, hi = lane >> 4;
  const int mBase = wid * 16;
  __shared__ float lcol[64];
  if (tid < 64) lcol[tid] = 0.f;
  __syncthreads();
  const __bf16* A  = ql + (size_t)bh * LL * DD;
  const __bf16* Bm = kl + (size_t)bh * LL * DD;
  f32x8 z[4];
#pragma unroll
  for (int nt = 0; nt < 4; ++nt) z[nt] = zero8();
#pragma unroll
  for (int ks = 0; ks < 2; ++ks) {
    bf16x16 a = frag_a(A, mBase + lo, DD, ks * 32, hi);
#pragma unroll
    for (int nt = 0; nt < 4; ++nt)
      z[nt] = wmma_bf16(a, frag_b(Bm, nt * 16 + lo, DD, ks * 32, hi), z[nt]);
  }
#pragma unroll
  for (int r = 0; r < 8; ++r) {
    float m = redmax16(fmaxf(fmaxf(z[0][r], z[1][r]), fmaxf(z[2][r], z[3][r])));
    float e[4], s = 0.f;
#pragma unroll
    for (int nt = 0; nt < 4; ++nt) { e[nt] = __expf(z[nt][r] - m); s += e[nt]; }
    s = redsum16(s);
    float inv = 1.f / s;
    int row = mBase + r + hi * 8;
#pragma unroll
    for (int nt = 0; nt < 4; ++nt) {
      float p = e[nt] * inv;
      k2[(size_t)bh * LL * LL + row * LL + nt * 16 + lo] = p;
      atomicAdd(&lcol[nt * 16 + lo], p);
    }
  }
  __syncthreads();
  if (tid < 64) atomicMax(gmax, __float_as_uint(lcol[tid]));
}

// ---------------- Newton-Schulz pseudo-inverse (f32 WMMA, all in LDS) --------
__device__ __forceinline__ void mm64_store(const float* A, const float* Bsrc,
                                           float alpha, bool useI,
                                           float* dst, float mul,
                                           int wid, int lo, int hi) {
  f32x8 acc[2];
#pragma unroll
  for (int t = 0; t < 2; ++t) acc[t] = zero8();
#pragma unroll
  for (int t = 0; t < 2; ++t) {
    int tile = wid * 2 + t;
    int mT = (tile >> 2) * 16, nT = (tile & 3) * 16;
#pragma unroll
    for (int ks = 0; ks < 16; ++ks) {
      int kb = ks * 4 + hi * 2;
      f32x2 a, b;
      a.x = A[(mT + lo) * 64 + kb];
      a.y = A[(mT + lo) * 64 + kb + 1];
      float b0 = Bsrc[kb * 64 + nT + lo];
      float b1 = Bsrc[(kb + 1) * 64 + nT + lo];
      if (useI) {
        b0 = ((kb == nT + lo) ? alpha : 0.f) - b0;
        b1 = ((kb + 1 == nT + lo) ? alpha : 0.f) - b1;
      }
      b.x = b0; b.y = b1;
      acc[t] = wmma_f32k4(a, b, acc[t]);
    }
  }
  __syncthreads();   // all reads done before (possibly in-place) store
#pragma unroll
  for (int t = 0; t < 2; ++t) {
    int tile = wid * 2 + t;
    int mT = (tile >> 2) * 16, nT = (tile & 3) * 16;
#pragma unroll
    for (int r = 0; r < 8; ++r)
      dst[(mT + r + hi * 8) * 64 + nT + lo] = acc[t][r] * mul;
  }
  __syncthreads();
}

__global__ __launch_bounds__(256)
void k_newton(const float* __restrict__ k2, const float* __restrict__ gmaxf,
              __bf16* __restrict__ viT) {
  const int bh = blockIdx.x;
  __shared__ float Km[4096], Vm[4096], T1[4096], T2[4096];
  const int tid = threadIdx.x, lane = tid & 31, wid = tid >> 5;
  const int lo = lane & 15, hi = lane >> 4;
  const float* src = k2 + (size_t)bh * 4096;
  for (int e = tid; e < 4096; e += 256) Km[e] = src[e];
  __syncthreads();
  const float inv = 1.f / *gmaxf;
  for (int e = tid; e < 4096; e += 256) {
    int i = e >> 6, j = e & 63;
    Vm[e] = Km[j * 64 + i] * inv;   // V = K^T / max(colsum)
  }
  __syncthreads();
  for (int it = 0; it < 6; ++it) {
    mm64_store(Km, Vm, 0.f,  false, T1, 1.f,   wid, lo, hi);  // T1 = K V
    mm64_store(T1, T1, 7.f,  true,  T2, 1.f,   wid, lo, hi);  // T2 = KV (7I - KV)
    mm64_store(T1, T2, 15.f, true,  T2, 1.f,   wid, lo, hi);  // T2 = KV (15I - T2)
    mm64_store(Vm, T2, 13.f, true,  Vm, 0.25f, wid, lo, hi);  // V  = 0.25 V (13I - T2)
  }
  for (int e = tid; e < 4096; e += 256) {
    int i = e >> 6, j = e & 63;
    viT[(size_t)bh * 4096 + j * 64 + i] = (__bf16)Vm[i * 64 + j];  // store V^-1 transposed
  }
}

// ---------------- kernel_3 @ v via flash-style online softmax over S ---------
__global__ __launch_bounds__(128)
void k_flash(const __bf16* __restrict__ ql, const __bf16* __restrict__ kbuf,
             const __bf16* __restrict__ vT, __bf16* __restrict__ k3vT) {
  const int bh = blockIdx.x;
  const int tid = threadIdx.x, lane = tid & 31, wid = tid >> 5;
  const int lo = lane & 15, hi = lane >> 4;
  const int mBase = wid * 16;
  __shared__ __align__(16) __bf16 lP[64 * 64];
  const __bf16* Aq   = ql + (size_t)bh * LL * DD;
  const __bf16* Ksrc = kbuf + (size_t)bh * SS * DD;
  const __bf16* Vt   = vT + (size_t)bh * DD * SS;
  bf16x16 aq[2];
  aq[0] = frag_a(Aq, mBase + lo, DD, 0, hi);
  aq[1] = frag_a(Aq, mBase + lo, DD, 32, hi);
  f32x8 o[4];
  float mrow[8], lrow[8];
#pragma unroll
  for (int nt = 0; nt < 4; ++nt) o[nt] = zero8();
#pragma unroll
  for (int r = 0; r < 8; ++r) { mrow[r] = -3.0e38f; lrow[r] = 0.f; }

  for (int sc = 0; sc < 64; ++sc) {
    const int s0 = sc * 64;
    f32x8 z[4];
#pragma unroll
    for (int nt = 0; nt < 4; ++nt) z[nt] = zero8();
#pragma unroll
    for (int ks = 0; ks < 2; ++ks)
#pragma unroll
      for (int nt = 0; nt < 4; ++nt)
        z[nt] = wmma_bf16(aq[ks], frag_b(Ksrc, s0 + nt * 16 + lo, DD, ks * 32, hi), z[nt]);

#pragma unroll
    for (int r = 0; r < 8; ++r) {
      float zm = redmax16(fmaxf(fmaxf(z[0][r], z[1][r]), fmaxf(z[2][r], z[3][r])));
      float mn = fmaxf(mrow[r], zm);
      float scl = __expf(mrow[r] - mn);
      float rs = 0.f;
      int row = mBase + r + hi * 8;
#pragma unroll
      for (int nt = 0; nt < 4; ++nt) {
        float p = __expf(z[nt][r] - mn);
        rs += p;
        lP[row * 64 + nt * 16 + lo] = (__bf16)p;
      }
      rs = redsum16(rs);
      lrow[r] = lrow[r] * scl + rs;
      mrow[r] = mn;
#pragma unroll
      for (int nt = 0; nt < 4; ++nt) o[nt][r] *= scl;
    }
    __syncthreads();
#pragma unroll
    for (int ks = 0; ks < 2; ++ks) {
      bf16x16 ap = frag_a(lP, mBase + lo, 64, ks * 32, hi);
#pragma unroll
      for (int nt = 0; nt < 4; ++nt)
        o[nt] = wmma_bf16(ap, frag_b(Vt, nt * 16 + lo, SS, s0 + ks * 32, hi), o[nt]);
    }
    __syncthreads();
  }
#pragma unroll
  for (int nt = 0; nt < 4; ++nt)
#pragma unroll
    for (int r = 0; r < 8; ++r) {
      int row = mBase + r + hi * 8, d = nt * 16 + lo;
      k3vT[(size_t)bh * 4096 + d * 64 + row] = (__bf16)(o[nt][r] / lrow[r]);
    }
}

// ---------------- fused: softmax(q k_l^T) -> @Vinv -> @k3v -> out ------------
__global__ __launch_bounds__(128)
void k_apply(const __bf16* __restrict__ qb, const __bf16* __restrict__ kl,
             const __bf16* __restrict__ viT, const __bf16* __restrict__ k3vT,
             float* __restrict__ out) {
  const int s0 = blockIdx.x * 64;
  const int bh = blockIdx.y;
  const int bidx = bh >> 4, h = bh & 15;
  const int tid = threadIdx.x, lane = tid & 31, wid = tid >> 5;
  const int lo = lane & 15, hi = lane >> 4;
  const int mBase = wid * 16;
  __shared__ __align__(16) __bf16 lP[64 * 64];
  __shared__ __align__(16) __bf16 lQ[64 * 64];
  const __bf16* Aq  = qb + (size_t)bh * SS * DD;
  const __bf16* Bkl = kl + (size_t)bh * LL * DD;
  const __bf16* Bvi = viT + (size_t)bh * 4096;
  const __bf16* Bk3 = k3vT + (size_t)bh * 4096;

  f32x8 z[4];
#pragma unroll
  for (int nt = 0; nt < 4; ++nt) z[nt] = zero8();
#pragma unroll
  for (int ks = 0; ks < 2; ++ks) {
    bf16x16 a = frag_a(Aq, s0 + mBase + lo, DD, ks * 32, hi);
#pragma unroll
    for (int nt = 0; nt < 4; ++nt)
      z[nt] = wmma_bf16(a, frag_b(Bkl, nt * 16 + lo, DD, ks * 32, hi), z[nt]);
  }
#pragma unroll
  for (int r = 0; r < 8; ++r) {
    float m = redmax16(fmaxf(fmaxf(z[0][r], z[1][r]), fmaxf(z[2][r], z[3][r])));
    float e[4], s = 0.f;
#pragma unroll
    for (int nt = 0; nt < 4; ++nt) { e[nt] = __expf(z[nt][r] - m); s += e[nt]; }
    s = redsum16(s);
    float inv = 1.f / s;
    int row = mBase + r + hi * 8;
#pragma unroll
    for (int nt = 0; nt < 4; ++nt) lP[row * 64 + nt * 16 + lo] = (__bf16)(e[nt] * inv);
  }
  __syncthreads();

  f32x8 a2[4];
#pragma unroll
  for (int nt = 0; nt < 4; ++nt) a2[nt] = zero8();
#pragma unroll
  for (int ks = 0; ks < 2; ++ks) {
    bf16x16 a = frag_a(lP, mBase + lo, 64, ks * 32, hi);
#pragma unroll
    for (int nt = 0; nt < 4; ++nt)
      a2[nt] = wmma_bf16(a, frag_b(Bvi, nt * 16 + lo, 64, ks * 32, hi), a2[nt]);
  }
#pragma unroll
  for (int nt = 0; nt < 4; ++nt)
#pragma unroll
    for (int r = 0; r < 8; ++r)
      lQ[(mBase + r + hi * 8) * 64 + nt * 16 + lo] = (__bf16)a2[nt][r];
  __syncthreads();

  f32x8 c[4];
#pragma unroll
  for (int nt = 0; nt < 4; ++nt) c[nt] = zero8();
#pragma unroll
  for (int ks = 0; ks < 2; ++ks) {
    bf16x16 a = frag_a(lQ, mBase + lo, 64, ks * 32, hi);
#pragma unroll
    for (int nt = 0; nt < 4; ++nt)
      c[nt] = wmma_bf16(a, frag_b(Bk3, nt * 16 + lo, 64, ks * 32, hi), c[nt]);
  }
#pragma unroll
  for (int nt = 0; nt < 4; ++nt)
#pragma unroll
    for (int r = 0; r < 8; ++r) {
      int s = s0 + mBase + r + hi * 8;
      int d = nt * 16 + lo;
      out[((size_t)bidx * SS + s) * DM + h * DD + d] = c[nt][r];
    }
}

// ---------------- depthwise conv over sequence, accumulate into out ----------
__global__ __launch_bounds__(256)
void k_conv(const __bf16* __restrict__ vT, const float* __restrict__ cw,
            float* __restrict__ out) {
  const int s0 = blockIdx.x * 64;
  const int bh = blockIdx.y;
  const int bidx = bh >> 4, h = bh & 15;
  __shared__ float lv[128 * 64];
  __shared__ float lw[CKSZ];
  const int tid = threadIdx.x;
  if (tid < CKSZ) lw[tid] = cw[h * CKSZ + tid];
  const __bf16* src = vT + (size_t)bh * DD * SS;
  for (int e = tid; e < 128 * 64; e += 256) {
    int d = e >> 7, sl = e & 127;
    int s = s0 - 32 + sl;
    float v = (s >= 0 && s < SS) ? (float)src[(size_t)d * SS + s] : 0.f;
    lv[sl * 64 + d] = v;
  }
  __syncthreads();
  const int d = tid & 63, r0 = tid >> 6;
#pragma unroll
  for (int ri = 0; ri < 16; ++ri) {
    int sl = r0 * 16 + ri;
    float acc = 0.f;
#pragma unroll
    for (int j = 0; j < CKSZ; ++j) acc += lv[(sl + j) * 64 + d] * lw[j];
    size_t oi = ((size_t)bidx * SS + s0 + sl) * DM + h * DD + d;
    out[oi] += acc;
  }
}

extern "C" void kernel_launch(void* const* d_in, const int* in_sizes, int n_in,
                              void* d_out, int out_size, void* d_ws, size_t ws_size,
                              hipStream_t stream) {
  (void)in_sizes; (void)n_in; (void)out_size; (void)ws_size;
  const float* X  = (const float*)d_in[0];
  const float* Wq = (const float*)d_in[1];
  const float* bq = (const float*)d_in[2];
  const float* Wk = (const float*)d_in[3];
  const float* bk = (const float*)d_in[4];
  const float* Wv = (const float*)d_in[5];
  const float* bv = (const float*)d_in[6];
  const float* cw = (const float*)d_in[7];
  char* ws = (char*)d_ws;
  __bf16* qb   = (__bf16*)(ws + 0);            // q bf16 [B*H, S, D]   (scaled)
  __bf16* kbuf = (__bf16*)(ws + 33554432);     // k bf16 [B*H, S, D]   (scaled)
  __bf16* vT   = (__bf16*)(ws + 67108864);     // v bf16 [B*H, D, S]
  __bf16* Xb   = (__bf16*)(ws + 100663296);    // hidden bf16 [B*S, DM]
  __bf16* Wt   = (__bf16*)(ws + 134217728);    // Wq|Wk|Wv bf16 transposed [n][k]
  __bf16* ql   = (__bf16*)(ws + 140509184);    // q_l bf16 [B*H, L, D]
  __bf16* kl   = (__bf16*)(ws + 141033472);    // k_l bf16 [B*H, L, D]
  float*  k2   = (float*) (ws + 141557760);    // kernel_2 f32 [B*H, L, L]
  __bf16* viT  = (__bf16*)(ws + 142606336);    // Vinv^T bf16 [B*H, L, L]
  __bf16* k3vT = (__bf16*)(ws + 143130624);    // (kernel_3 @ v)^T bf16 [B*H, D, L]
  unsigned* gmax = (unsigned*)(ws + 143654912);
  float* out = (float*)d_out;

  k_cvt_x<<<8192, 256, 0, stream>>>(X, Xb);
  k_cvt_wt<<<dim3(32, 32, 3), 256, 0, stream>>>(Wq, Wk, Wv, Wt);
  k_zero_scalar<<<1, 32, 0, stream>>>(gmax);
  k_qkv<<<dim3(8, 128, 3), 256, 0, stream>>>(Xb, Wt, bq, bk, bv, qb, kbuf, vT);
  k_landmarks<<<BB * HH * LL, 64, 0, stream>>>(qb, kbuf, ql, kl);
  k_kernel2<<<BB * HH, 128, 0, stream>>>(ql, kl, k2, gmax);
  k_newton<<<BB * HH, 256, 0, stream>>>(k2, (const float*)gmax, viT);
  k_flash<<<BB * HH, 128, 0, stream>>>(ql, kbuf, vT, k3vT);
  k_apply<<<dim3(64, 64), 128, 0, stream>>>(qb, kl, viT, k3vT, out);
  k_conv<<<dim3(64, 64), 256, 0, stream>>>(vT, cw, out);
}
